// GhostLinear_15573551415405
// MI455X (gfx1250) — matrix operand order repaired and checked by reading, not compiled, not run on hardware
//
#include <hip/hip_runtime.h>

// GhostLinear on MI455X (gfx1250):
//   W = lut[indices] * scale   (4096 x 4096, 256-code LUT)
//   out[16384, 4096] = x[16384, 4096] @ W^T
// bf16 WMMA GEMM (v_wmma_f32_16x16x32_bf16) with f32 accumulation, fed by
// gfx1250 async global->LDS copies (ASYNCcnt) -- no VGPR staging, no spills.

typedef __attribute__((ext_vector_type(16))) __bf16 v16bf;
typedef __attribute__((ext_vector_type(8)))  float  v8f;

#define IN_DIM   4096
#define OUT_DIM  4096
#define M_DIM    16384
#define BM 128
#define BN 128
#define BK 32
#define BKP 40          // padded LDS row stride (elements); 80B, 16B-aligned

// float -> bf16, round-to-nearest-even
__device__ __forceinline__ __bf16 f2bf(float f) {
    union { float f; unsigned u; } x; x.f = f;
    unsigned r = x.u + 0x7FFFu + ((x.u >> 16) & 1u);
    unsigned short h = (unsigned short)(r >> 16);
    union { unsigned short s; __bf16 b; } y; y.s = h;
    return y.b;
}

// gfx1250 async copy: 16B global -> LDS, tracked by ASYNCcnt.
// lds = low 32 bits of the generic shared-memory address (ISA: generic->LDS
// mapping truncates to addr[31:0]); gaddr = 64-bit global address (VGPR pair).
__device__ __forceinline__ void async_ld_b128(unsigned lds, const void* gaddr) {
    asm volatile("global_load_async_to_lds_b128 %0, %1, off"
                 :: "v"(lds), "v"((unsigned long long)(uintptr_t)gaddr)
                 : "memory");
}

__device__ __forceinline__ void wait_async0() {
#if __has_builtin(__builtin_amdgcn_s_wait_asynccnt)
    __builtin_amdgcn_s_wait_asynccnt(0);
#else
    asm volatile("s_wait_asynccnt 0x0" ::: "memory");
#endif
}

// ---------------------------------------------------------------------------
// Kernel 1: W_bf16[o][i] = bf16(lut[idx[o][i]] * scale[o]); LUT staged in LDS.
// ---------------------------------------------------------------------------
__global__ void __launch_bounds__(256)
ghost_dequant_w(const int* __restrict__ idx, const float* __restrict__ lut,
                const float* __restrict__ scale, __bf16* __restrict__ wq) {
    __shared__ float slut[256];
    const int tid = threadIdx.x;
    slut[tid] = lut[tid];
    __syncthreads();

    const size_t e0 = ((size_t)blockIdx.x * 256 + tid) * 8;   // aligned, divides rows
    const int o = (int)(e0 >> 12);                            // e0 / 4096
    const float s = scale[o];

    const int4* ip = (const int4*)(idx + e0);
    int4 i0 = ip[0], i1 = ip[1];
    int ii[8] = { i0.x, i0.y, i0.z, i0.w, i1.x, i1.y, i1.z, i1.w };

    union { __bf16 b[8]; uint4 u; } outp;
#pragma unroll
    for (int j = 0; j < 8; ++j) outp.b[j] = f2bf(slut[ii[j] & 255] * s);
    *(uint4*)(wq + e0) = outp.u;
}

// ---------------------------------------------------------------------------
// Kernel 2: x fp32 -> bf16 (8 elements / thread)
// ---------------------------------------------------------------------------
__global__ void __launch_bounds__(256)
ghost_cvt_x(const float* __restrict__ x, __bf16* __restrict__ xq) {
    const size_t e0 = ((size_t)blockIdx.x * 256 + threadIdx.x) * 8;
    float4 a = *(const float4*)(x + e0);
    float4 b = *(const float4*)(x + e0 + 4);
    union { __bf16 b[8]; uint4 u; } o;
    o.b[0] = f2bf(a.x); o.b[1] = f2bf(a.y); o.b[2] = f2bf(a.z); o.b[3] = f2bf(a.w);
    o.b[4] = f2bf(b.x); o.b[5] = f2bf(b.y); o.b[6] = f2bf(b.z); o.b[7] = f2bf(b.w);
    *(uint4*)(xq + e0) = o.u;
}

// ---------------------------------------------------------------------------
// Kernel 3: bf16 WMMA GEMM.  out[M,N] = Xq[M,K] * Wq[N,K]^T
// 128x128 block tile, BK=32, 8 waves (wave tile 32x64 = 2x4 WMMA tiles),
// double-buffered LDS filled by async global->LDS b128 copies.
// ---------------------------------------------------------------------------
__global__ void __launch_bounds__(256)
ghost_wmma_gemm(const __bf16* __restrict__ Xq, const __bf16* __restrict__ Wq,
                float* __restrict__ out) {
    __shared__ __align__(16) __bf16 Ash[2][BM][BKP];
    __shared__ __align__(16) __bf16 Bsh[2][BN][BKP];

    const int tid  = threadIdx.x;
    const int lane = tid & 31;
    const int wave = tid >> 5;
    const int bn0  = blockIdx.x * BN;
    const int bm0  = blockIdx.y * BM;

    // cooperative tile fill: 2 threads per 128-row, 32B (2 x async b128) each
    const int lrow  = tid >> 1;
    const int lhalf = tid & 1;
    const __bf16* gA = Xq + (size_t)(bm0 + lrow) * IN_DIM + lhalf * 16;
    const __bf16* gB = Wq + (size_t)(bn0 + lrow) * IN_DIM + lhalf * 16;

    // per-buffer LDS byte addresses for this thread's fill slot
    unsigned ldsA[2], ldsB[2];
#pragma unroll
    for (int b = 0; b < 2; ++b) {
        ldsA[b] = (unsigned)(uintptr_t)&Ash[b][lrow][lhalf * 16];
        ldsB[b] = (unsigned)(uintptr_t)&Bsh[b][lrow][lhalf * 16];
    }

    // wave tiling: 4 waves along M, 2 along N
    const int wm = (wave >> 1) * 32;
    const int wn = (wave &  1) * 64;
    const int lg = lane >> 4;     // lane group (ISA K-half select)
    const int lr = lane & 15;     // row (A) / col (B) within 16

    v8f zero = {};
    v8f acc[2][4];
#pragma unroll
    for (int i = 0; i < 2; ++i)
#pragma unroll
        for (int j = 0; j < 4; ++j) acc[i][j] = zero;

    // async fill of buffer 0
    async_ld_b128(ldsA[0],      gA);
    async_ld_b128(ldsA[0] + 16, gA + 8);
    async_ld_b128(ldsB[0],      gB);
    async_ld_b128(ldsB[0] + 16, gB + 8);
    wait_async0();
    __syncthreads();

    const int KT = IN_DIM / BK;   // 128 k-tiles

#pragma unroll 1
    for (int kt = 0; kt < KT; ++kt) {
        const int cur = kt & 1;

        // kick off async copy of the next tile into the other buffer;
        // it lands in LDS while this iteration's WMMAs run.
        if (kt + 1 < KT) {
            const __bf16* nA = gA + (size_t)(kt + 1) * BK;
            const __bf16* nB = gB + (size_t)(kt + 1) * BK;
            async_ld_b128(ldsA[cur ^ 1],      nA);
            async_ld_b128(ldsA[cur ^ 1] + 16, nA + 8);
            async_ld_b128(ldsB[cur ^ 1],      nB);
            async_ld_b128(ldsB[cur ^ 1] + 16, nB + 8);
        }

        // ---- compute on buf[cur]: 2 A frags, 4 B frags, 8 WMMAs ----
        union FragU { v16bf v; uint4 u[2]; };
        FragU a[2], b[4];
        const int kba = lg * 8;    // A: K chunks at +0 and +16 (ISA A layout)
        const int kbb = lg * 8;    // B: K chunks at +0 and +16 (same as A)
#pragma unroll
        for (int mt = 0; mt < 2; ++mt) {
            a[mt].u[0] = *(const uint4*)&Ash[cur][wm + mt * 16 + lr][kba];
            a[mt].u[1] = *(const uint4*)&Ash[cur][wm + mt * 16 + lr][kba + 16];
        }
#pragma unroll
        for (int nt = 0; nt < 4; ++nt) {
            b[nt].u[0] = *(const uint4*)&Bsh[cur][wn + nt * 16 + lr][kbb];
            b[nt].u[1] = *(const uint4*)&Bsh[cur][wn + nt * 16 + lr][kbb + 16];
        }
#pragma unroll
        for (int mt = 0; mt < 2; ++mt)
#pragma unroll
            for (int nt = 0; nt < 4; ++nt)
                acc[mt][nt] = __builtin_amdgcn_wmma_f32_16x16x32_bf16(
                    false, a[mt].v, false, b[nt].v,
                    (short)0, acc[mt][nt], false, false);

        // this wave's async writes must have landed before the barrier
        wait_async0();
        __syncthreads();
    }

    // ---- store: C/D layout: vgpr r -> m = 8*(lane>>4) + r, n = lane&15 ----
#pragma unroll
    for (int mt = 0; mt < 2; ++mt) {
#pragma unroll
        for (int nt = 0; nt < 4; ++nt) {
            const int m0 = bm0 + wm + mt * 16 + lg * 8;
            const int n  = bn0 + wn + nt * 16 + lr;
#pragma unroll
            for (int r = 0; r < 8; ++r)
                out[(size_t)(m0 + r) * OUT_DIM + n] = acc[mt][nt][r];
        }
    }
}

// ---------------------------------------------------------------------------
extern "C" void kernel_launch(void* const* d_in, const int* in_sizes, int n_in,
                              void* d_out, int out_size, void* d_ws, size_t ws_size,
                              hipStream_t stream) {
    const float* x       = (const float*)d_in[0];   // [8,2048,4096] fp32
    const float* scale   = (const float*)d_in[1];   // [4096,1] fp32
    const float* lut     = (const float*)d_in[2];   // [256] fp32
    const int*   indices = (const int*)d_in[3];     // [4096,4096] int32
    float* out = (float*)d_out;                     // [8,2048,4096] fp32

    __bf16* wq = (__bf16*)d_ws;                                         // 32 MB
    __bf16* xq = (__bf16*)((char*)d_ws + (size_t)OUT_DIM * IN_DIM * 2); // 134 MB

    ghost_dequant_w<<<(OUT_DIM * (size_t)IN_DIM) / (256 * 8), 256, 0, stream>>>(
        indices, lut, scale, wq);
    ghost_cvt_x<<<((size_t)M_DIM * IN_DIM) / (256 * 8), 256, 0, stream>>>(x, xq);
    dim3 grid(OUT_DIM / BN, M_DIM / BM);   // (32, 128)
    ghost_wmma_gemm<<<grid, 256, 0, stream>>>(xq, wq, out);
}
